// GCN_75909251990135
// MI455X (gfx1250) — matrix operand order repaired and checked by reading, not compiled, not run on hardware
//
#include <hip/hip_runtime.h>
#include <hip/hip_bf16.h>

// Problem constants (match reference)
#define NN   200000
#define EE   400000
#define GG   4096
#define FIN  960
#define HH   128
#define LLAY 3
#define EPSF 1e-5f

typedef __bf16 bf16_t;
typedef bf16_t v16bf __attribute__((ext_vector_type(16)));
typedef float  v8f   __attribute__((ext_vector_type(8)));

// ---------------------------------------------------------------------------
// WMMA fragment loads from LDS (stride 34 bf16 elements per row).
// A 16x32 bf16 layout: lanes 0-15 = M rows, K 0..15;
// lanes 16-31 = same M rows, K 16..31.
// B 32x16 bf16: lane L<16 = col N=L with K 0..15; lane L+16 same col, K 16..31.
// Bs is stored transposed ([n][k]) so both are contiguous per-lane reads.
// ---------------------------------------------------------------------------
__device__ __forceinline__ v16bf frag_a(const bf16_t* Ls, int row, int half) {
  const bf16_t* p = Ls + row * 34;
  const int o0 = half * 16, o1 = half * 16 + 8;
  v16bf v;
#pragma unroll
  for (int i = 0; i < 8; ++i) v[i] = p[o0 + i];
#pragma unroll
  for (int i = 0; i < 8; ++i) v[8 + i] = p[o1 + i];
  return v;
}
__device__ __forceinline__ v16bf frag_b(const bf16_t* Ls, int row, int half) {
  const bf16_t* p = Ls + row * 34 + half * 16;
  v16bf v;
#pragma unroll
  for (int i = 0; i < 16; ++i) v[i] = p[i];
  return v;
}

// ---------------------------------------------------------------------------
// C[M x 128] = A[M x K](f32, cvt->bf16 on the fly) @ B[K x 128](bf16) + bias
// Block tile 128x128, 8 waves, each wave = one 16-row strip x 8 N-tiles.
// ---------------------------------------------------------------------------
__global__ __launch_bounds__(256)
void wmma_gemm_bf16(const float* __restrict__ A, int M, int K,
                    const bf16_t* __restrict__ B,
                    const float* __restrict__ bias, int relu,
                    float* __restrict__ C) {
  __shared__ bf16_t As[128 * 34];
  __shared__ bf16_t Bs[128 * 34];
  const int t    = threadIdx.x;
  const int lane = t & 31;
  const int wave = t >> 5;
  const int half = lane >> 4;
  const int lr   = lane & 15;
  const int m0   = blockIdx.x * 128;

  v8f acc[8];
#pragma unroll
  for (int i = 0; i < 8; ++i) acc[i] = (v8f)0.0f;

  for (int k0 = 0; k0 < K; k0 += 32) {
    // Stage A tile: 128 rows x 32 cols fp32 -> bf16 (4 float4 per thread)
#pragma unroll
    for (int i = 0; i < 4; ++i) {
      int p  = t + i * 256;           // 0..1023 float4 slots
      int r  = p >> 3;
      int c4 = (p & 7) * 4;
      int gm = m0 + r;
      float4 v = make_float4(0.f, 0.f, 0.f, 0.f);
      if (gm < M) v = *(const float4*)(A + (size_t)gm * K + k0 + c4);
      bf16_t* d = &As[r * 34 + c4];
      d[0] = (bf16_t)v.x; d[1] = (bf16_t)v.y;
      d[2] = (bf16_t)v.z; d[3] = (bf16_t)v.w;
    }
    // Stage B tile transposed: Bs[n][k] <- B[k0+k][n]
    {
      int kk = t >> 3;                // 0..31
      int n0 = (t & 7) * 16;
      const bf16_t* src = B + (size_t)(k0 + kk) * HH + n0;
#pragma unroll
      for (int i = 0; i < 16; ++i) Bs[(n0 + i) * 34 + kk] = src[i];
    }
    __syncthreads();

    v16bf a = frag_a(As, wave * 16 + lr, half);
#pragma unroll
    for (int nt = 0; nt < 8; ++nt) {
      v16bf b = frag_b(Bs, nt * 16 + lr, half);
      acc[nt] = __builtin_amdgcn_wmma_f32_16x16x32_bf16(
          false, a, false, b, (short)0, acc[nt], false, false);
    }
    __syncthreads();
  }

  // Store: lane L<16 -> col lr, rows r..r+7; lane>=16 -> rows 8..15
#pragma unroll
  for (int nt = 0; nt < 8; ++nt) {
    int n = nt * 16 + lr;
    float bv = bias ? bias[n] : 0.0f;
#pragma unroll
    for (int r = 0; r < 8; ++r) {
      int gm = m0 + wave * 16 + half * 8 + r;
      if (gm < M) {
        float v = acc[nt][r] + bv;
        if (relu) v = fmaxf(v, 0.0f);
        C[(size_t)gm * HH + n] = v;
      }
    }
  }
}

// ---------------------------------------------------------------------------
// Elementwise / graph kernels (fp32, atomic paths)
// ---------------------------------------------------------------------------
__global__ void k_zero(float* p, int n) {
  int i = blockIdx.x * 256 + threadIdx.x;
  if (i < n) p[i] = 0.0f;
}
__global__ void k_fill(float* p, float v, int n) {
  int i = blockIdx.x * 256 + threadIdx.x;
  if (i < n) p[i] = v;
}
__global__ void k_cvt_bf16(const float* __restrict__ s, bf16_t* __restrict__ d, int n) {
  int i = blockIdx.x * 256 + threadIdx.x;
  if (i < n) d[i] = (bf16_t)s[i];
}
__global__ void k_deg(const int* __restrict__ col, float* deg, int e) {
  int i = blockIdx.x * 256 + threadIdx.x;
  if (i < e) atomicAdd(&deg[col[i]], 1.0f);
}
__global__ void k_dinv(float* d, int n) {
  int i = blockIdx.x * 256 + threadIdx.x;
  if (i < n) d[i] = rsqrtf(1.0f + d[i]);
}
__global__ void k_cnt(const int* __restrict__ batch, float* cnt, int n) {
  int i = blockIdx.x * 256 + threadIdx.x;
  if (i < n) atomicAdd(&cnt[batch[i]], 1.0f);
}
// h[n][f] = dinv[n]^2 * hw[n][f] + b_l[f]   (float4 over N*H/4)
__global__ void k_selfbias(const float* __restrict__ dinv, const float* __restrict__ hw,
                           const float* __restrict__ bl, float* __restrict__ h, int n4) {
  int i = blockIdx.x * 256 + threadIdx.x;
  if (i >= n4) return;
  int n = i >> 5, q = (i & 31) * 4;
  float d2 = dinv[n] * dinv[n];
  float4 v = *(const float4*)(hw + (size_t)n * HH + q);
  float4 o;
  o.x = d2 * v.x + bl[q + 0]; o.y = d2 * v.y + bl[q + 1];
  o.z = d2 * v.z + bl[q + 2]; o.w = d2 * v.w + bl[q + 3];
  *(float4*)(h + (size_t)n * HH + q) = o;
}
// one wave per edge, lane covers 4 features
__global__ void k_edge_agg(const int* __restrict__ er, const int* __restrict__ ec,
                           const float* __restrict__ dinv, const float* __restrict__ hw,
                           float* __restrict__ out, int e) {
  int eid = blockIdx.x * 8 + (threadIdx.x >> 5);
  if (eid >= e) return;
  int lane = threadIdx.x & 31;
  int r = er[eid], c = ec[eid];
  float nm = dinv[r] * dinv[c];
  float4 v = *(const float4*)(hw + (size_t)r * HH + lane * 4);
  float* o = out + (size_t)c * HH + lane * 4;
  atomicAdd(o + 0, nm * v.x); atomicAdd(o + 1, nm * v.y);
  atomicAdd(o + 2, nm * v.z); atomicAdd(o + 3, nm * v.w);
}
__global__ void k_gn_sum(const float* __restrict__ h, const int* __restrict__ batch,
                         float* __restrict__ mean, int n4) {
  int i = blockIdx.x * 256 + threadIdx.x;
  if (i >= n4) return;
  int n = i >> 5, q = (i & 31) * 4;
  int g = batch[n];
  float4 v = *(const float4*)(h + (size_t)n * HH + q);
  float* m = mean + (size_t)g * HH + q;
  atomicAdd(m + 0, v.x); atomicAdd(m + 1, v.y);
  atomicAdd(m + 2, v.z); atomicAdd(m + 3, v.w);
}
__global__ void k_div_cnt(float* __restrict__ m, const float* __restrict__ cnt, int gh) {
  int i = blockIdx.x * 256 + threadIdx.x;
  if (i < gh) m[i] /= fmaxf(cnt[i >> 7], 1.0f);
}
__global__ void k_gn_var(const float* __restrict__ h, const int* __restrict__ batch,
                         const float* __restrict__ mean, const float* __restrict__ a,
                         float* __restrict__ var, int n4) {
  int i = blockIdx.x * 256 + threadIdx.x;
  if (i >= n4) return;
  int n = i >> 5, q = (i & 31) * 4;
  int g = batch[n];
  const float* hp = h + (size_t)n * HH + q;
  const float* mp = mean + (size_t)g * HH + q;
  float* vp = var + (size_t)g * HH + q;
#pragma unroll
  for (int j = 0; j < 4; ++j) {
    float cdiff = hp[j] - a[q + j] * mp[j];
    atomicAdd(vp + j, cdiff * cdiff);
  }
}
__global__ void k_invstd(float* __restrict__ v, const float* __restrict__ cnt, int gh) {
  int i = blockIdx.x * 256 + threadIdx.x;
  if (i < gh) v[i] = rsqrtf(v[i] / fmaxf(cnt[i >> 7], 1.0f) + EPSF);
}
__global__ void k_gn_apply(float* __restrict__ h, const int* __restrict__ batch,
                           const float* __restrict__ mean, const float* __restrict__ istd,
                           const float* __restrict__ a, const float* __restrict__ w,
                           const float* __restrict__ b, int n4) {
  int i = blockIdx.x * 256 + threadIdx.x;
  if (i >= n4) return;
  int n = i >> 5, q = (i & 31) * 4;
  int g = batch[n];
  float* hp = h + (size_t)n * HH + q;
  const float* mp = mean + (size_t)g * HH + q;
  const float* sp = istd + (size_t)g * HH + q;
#pragma unroll
  for (int j = 0; j < 4; ++j) {
    float c = hp[j] - a[q + j] * mp[j];
    hp[j] = fmaxf(c * sp[j] * w[q + j] + b[q + j], 0.0f);
  }
}
// gate[n] = dot(gh[n,:], g2w) + g2b ; one wave per node
__global__ void k_gate(const float* __restrict__ gh, const float* __restrict__ g2w,
                       const float* __restrict__ g2b, float* __restrict__ gate, int n) {
  int node = blockIdx.x * 8 + (threadIdx.x >> 5);
  if (node >= n) return;
  int lane = threadIdx.x & 31;
  float4 hv = *(const float4*)(gh + (size_t)node * HH + lane * 4);
  float4 wv = *(const float4*)(g2w + lane * 4);
  float v = hv.x * wv.x + hv.y * wv.y + hv.z * wv.z + hv.w * wv.w;
#pragma unroll
  for (int off = 16; off; off >>= 1) v += __shfl_down(v, off, 32);
  if (lane == 0) gate[node] = v + g2b[0];
}
__device__ __forceinline__ void atomicMaxF(float* addr, float val) {
  unsigned int* ua = (unsigned int*)addr;
  unsigned int old = *ua;
  while (__uint_as_float(old) < val) {
    unsigned int assumed = old;
    old = atomicCAS(ua, assumed, __float_as_uint(val));
    if (old == assumed) break;
  }
}
__global__ void k_amax(const float* __restrict__ gate, const int* __restrict__ batch,
                       float* __restrict__ mg, int n) {
  int i = blockIdx.x * 256 + threadIdx.x;
  if (i < n) atomicMaxF(&mg[batch[i]], gate[i]);
}
__global__ void k_expsum(float* __restrict__ gate, const int* __restrict__ batch,
                         const float* __restrict__ mg, float* __restrict__ sg, int n) {
  int i = blockIdx.x * 256 + threadIdx.x;
  if (i >= n) return;
  int g = batch[i];
  float e = __expf(gate[i] - mg[g]);
  gate[i] = e;
  atomicAdd(&sg[g], e);
}
__global__ void k_attn(const float* __restrict__ gate, const int* __restrict__ batch,
                       const float* __restrict__ sg, const float* __restrict__ h,
                       float* __restrict__ out, int n4) {
  int i = blockIdx.x * 256 + threadIdx.x;
  if (i >= n4) return;
  int n = i >> 5, q = (i & 31) * 4;
  int g = batch[n];
  float w = gate[n] / fmaxf(sg[g], 1e-16f);
  float4 v = *(const float4*)(h + (size_t)n * HH + q);
  float* o = out + (size_t)g * HH + q;
  atomicAdd(o + 0, w * v.x); atomicAdd(o + 1, w * v.y);
  atomicAdd(o + 2, w * v.z); atomicAdd(o + 3, w * v.w);
}

// ---------------------------------------------------------------------------
extern "C" void kernel_launch(void* const* d_in, const int* in_sizes, int n_in,
                              void* d_out, int out_size, void* d_ws, size_t ws_size,
                              hipStream_t stream) {
  const float* x     = (const float*)d_in[0];
  const int*   ei    = (const int*)d_in[1];   // harness passes integer inputs as int32
  const int*   batch = (const int*)d_in[2];
  const float* W_in  = (const float*)d_in[3];
  const float* b_in  = (const float*)d_in[4];
  const float* W_gcn = (const float*)d_in[5];
  const float* b_gcn = (const float*)d_in[6];
  const float* gn_w  = (const float*)d_in[7];
  const float* gn_b  = (const float*)d_in[8];
  const float* gn_a  = (const float*)d_in[9];
  const float* g1_w  = (const float*)d_in[10];
  const float* g1_b  = (const float*)d_in[11];
  const float* g2_w  = (const float*)d_in[12];
  const float* g2_b  = (const float*)d_in[13];
  const int* erow = ei;
  const int* ecol = ei + EE;

  // Workspace carve (~210 MB total)
  float* ws   = (float*)d_ws;
  float* h    = ws;                         // N*H
  float* hw   = h   + (size_t)NN * HH;      // N*H
  float* dinv = hw  + (size_t)NN * HH;      // N
  float* cnt  = dinv + NN;                  // G
  float* mean = cnt  + GG;                  // G*H
  float* var  = mean + (size_t)GG * HH;     // G*H
  float* gate = var  + (size_t)GG * HH;     // N
  float* mg   = gate + NN;                  // G
  float* sg   = mg   + GG;                  // G
  bf16_t* Wbf = (bf16_t*)(sg + GG);         // FIN*H bf16 (weight staging)
  float* out  = (float*)d_out;              // G*H

  const int NH4 = NN * (HH / 4);            // 6.4M
  const int GH  = GG * HH;                  // 524288
  const int TB  = 256;
  auto blk = [](int n) { return (n + 255) / 256; };

  // degree -> dinv, per-graph counts
  k_zero<<<blk(NN), TB, 0, stream>>>(dinv, NN);
  k_zero<<<blk(GG), TB, 0, stream>>>(cnt, GG);
  k_deg<<<blk(EE), TB, 0, stream>>>(ecol, dinv, EE);
  k_dinv<<<blk(NN), TB, 0, stream>>>(dinv, NN);
  k_cnt<<<blk(NN), TB, 0, stream>>>(batch, cnt, NN);

  // h = x @ W_in + b_in
  k_cvt_bf16<<<blk(FIN * HH), TB, 0, stream>>>(W_in, Wbf, FIN * HH);
  wmma_gemm_bf16<<<(NN + 127) / 128, TB, 0, stream>>>(x, NN, FIN, Wbf, b_in, 0, h);

  for (int l = 0; l < LLAY; ++l) {
    // hw = h @ W_gcn[l]
    k_cvt_bf16<<<blk(HH * HH), TB, 0, stream>>>(W_gcn + (size_t)l * HH * HH, Wbf, HH * HH);
    wmma_gemm_bf16<<<(NN + 127) / 128, TB, 0, stream>>>(h, NN, HH, Wbf, nullptr, 0, hw);
    // h = self_w * hw + b ; then scatter-add edges
    k_selfbias<<<blk(NH4), TB, 0, stream>>>(dinv, hw, b_gcn + (size_t)l * HH, h, NH4);
    k_edge_agg<<<(EE + 7) / 8, TB, 0, stream>>>(erow, ecol, dinv, hw, h, EE);
    // GraphNorm + ReLU
    k_zero<<<blk(GH), TB, 0, stream>>>(mean, GH);
    k_zero<<<blk(GH), TB, 0, stream>>>(var, GH);
    k_gn_sum<<<blk(NH4), TB, 0, stream>>>(h, batch, mean, NH4);
    k_div_cnt<<<blk(GH), TB, 0, stream>>>(mean, cnt, GH);
    k_gn_var<<<blk(NH4), TB, 0, stream>>>(h, batch, mean, gn_a + (size_t)l * HH, var, NH4);
    k_invstd<<<blk(GH), TB, 0, stream>>>(var, cnt, GH);
    k_gn_apply<<<blk(NH4), TB, 0, stream>>>(h, batch, mean, var,
                                            gn_a + (size_t)l * HH,
                                            gn_w + (size_t)l * HH,
                                            gn_b + (size_t)l * HH, NH4);
  }

  // Attentional aggregation
  k_cvt_bf16<<<blk(HH * HH), TB, 0, stream>>>(g1_w, Wbf, HH * HH);
  wmma_gemm_bf16<<<(NN + 127) / 128, TB, 0, stream>>>(h, NN, HH, Wbf, g1_b, 1, hw);
  k_gate<<<(NN + 7) / 8, TB, 0, stream>>>(hw, g2_w, g2_b, gate, NN);
  k_fill<<<blk(GG), TB, 0, stream>>>(mg, -1e30f, GG);
  k_zero<<<blk(GG), TB, 0, stream>>>(sg, GG);
  k_zero<<<blk(GH), TB, 0, stream>>>(out, GH);
  k_amax<<<blk(NN), TB, 0, stream>>>(gate, batch, mg, NN);
  k_expsum<<<blk(NN), TB, 0, stream>>>(gate, batch, mg, sg, NN);
  k_attn<<<blk(NH4), TB, 0, stream>>>(gate, batch, sg, h, out, NH4);
}
